// Encoder_70643622085260
// MI455X (gfx1250) — matrix operand order repaired and checked by reading, not compiled, not run on hardware
//
#include <hip/hip_runtime.h>

typedef __attribute__((ext_vector_type(16))) __bf16 v16bf;
typedef __attribute__((ext_vector_type(8)))  __bf16 v8bf;
typedef __attribute__((ext_vector_type(8)))  float  v8f;

union V16 { v16bf v; v8bf h[2]; };

__device__ inline v8bf zero8() {
  v8bf z;
#pragma unroll
  for (int i = 0; i < 8; ++i) z[i] = (__bf16)0.0f;
  return z;
}

// ---------------- mask prep ----------------
__global__ __launch_bounds__(256) void mask_from_bool(const unsigned char* mb, float* m0, int n) {
  int i = blockIdx.x * 256 + threadIdx.x;
  if (i < n) m0[i] = mb[i] ? 1.0f : 0.0f;
}

// occupancy pooling: dst dim = Ds/2, dst[v] = max over 2^3 children
__global__ __launch_bounds__(256) void pool_mask(const float* src, float* dst, int Ds) {
  int Dd = Ds >> 1;
  int nd = Dd * Dd * Dd;
  int total = 2 * nd;
  int i = blockIdx.x * 256 + threadIdx.x;
  if (i >= total) return;
  int b = i / nd, v = i % nd;
  int z = v / (Dd * Dd), y = (v / Dd) % Dd, x = v % Dd;
  const float* s = src + (size_t)b * Ds * Ds * Ds;
  float mx = 0.0f;
#pragma unroll
  for (int dz = 0; dz < 2; ++dz)
#pragma unroll
    for (int dy = 0; dy < 2; ++dy)
#pragma unroll
      for (int dx = 0; dx < 2; ++dx)
        mx = fmaxf(mx, s[((z * 2 + dz) * Ds + (y * 2 + dy)) * Ds + (x * 2 + dx)]);
  dst[i] = mx;
}

// ---------------- input prep: x*m0 -> channels-last bf16, Cin padded to 32 ----------------
__global__ __launch_bounds__(256) void prep_input(const float* x, const float* m0, __bf16* h) {
  const int NV = 64 * 64 * 64;
  int i = blockIdx.x * 256 + threadIdx.x;   // over 2*NV voxels
  if (i >= 2 * NV) return;
  int b = i / NV, v = i % NV;
  float m = m0[i];
  __bf16* o = h + (size_t)i * 32;
  const float* xb = x + (size_t)b * 3 * NV + v;
#pragma unroll
  for (int c = 0; c < 32; ++c) {
    float val = (c < 3) ? xb[(size_t)c * NV] * m : 0.0f;
    o[c] = (__bf16)val;
  }
}

// ---------------- weight pack: [Cout][Cin][27] f32 -> [27][Cout][CinPad] bf16 ----------------
__global__ __launch_bounds__(256) void wt_pack(const float* w, __bf16* wt,
                                               int Cout, int Cin, int CinPad, int total) {
  int i = blockIdx.x * 256 + threadIdx.x;
  if (i >= total) return;
  int ci  = i % CinPad;
  int t   = i / CinPad;
  int co  = t % Cout;
  int tap = t / Cout;
  float v = (ci < Cin) ? w[((size_t)co * Cin + ci) * 27 + tap] : 0.0f;
  wt[i] = (__bf16)v;
}

// ---------------- implicit-GEMM masked conv3d via WMMA bf16 ----------------
// in  : [B][Din^3][CIN] bf16 (already masked)   wt : [27][COUT][CIN] bf16
// one wave: 16 output voxels x 64 output channels (4 C tiles).
// Per-tap weight tile (COUT x CIN) async-staged into LDS (double buffered),
// shared by all 8 waves of the block. CIN/COUT are compile-time so the
// staging and K loops fully unroll.
template <int CIN, int COUT>
__global__ __launch_bounds__(256) void conv3d_wmma(
    const __bf16* __restrict__ in, const __bf16* __restrict__ wt,
    const float* __restrict__ bias, const float* __restrict__ mask,
    const __bf16* __restrict__ res, __bf16* __restrict__ out,
    float* __restrict__ outF,
    int Din, int Dout, int stride, int relu) {
  constexpr int E      = COUT * CIN;       // bf16 elements per tap weight tile
  constexpr int NCOPY  = E / 2048;         // 16B copies per thread per tap
  constexpr int KCHUNK = CIN / 32;

  __shared__ __attribute__((aligned(16))) __bf16 smem[2][E];

  const int tid  = threadIdx.x;
  const int lane = tid & 31;
  const int wave = blockIdx.x * 8 + (tid >> 5);

  const int NVo = Dout * Dout * Dout;
  const int NVi = Din * Din * Din;
  constexpr int coutTiles = COUT >> 6;     // 64 couts per wave
  const int voxTiles = NVo >> 4;

  int t  = wave;
  const int ct = t % coutTiles; t /= coutTiles;
  const int vt = t % voxTiles;
  const int b  = t / voxTiles;

  const int m  = lane & 15;                // voxel row within tile (A/C layout)
  const int hi = lane >> 4;                // K-half selector
  const int v  = vt * 16 + m;
  const int oz = v / (Dout * Dout);
  const int oy = (v / Dout) % Dout;
  const int ox = v % Dout;
  const int co0 = ct * 64;

  const __bf16* inB = in + (size_t)b * NVi * CIN;

  // async-stage tap weights into LDS buffer nb (16B per lane per issue)
  auto stage = [&](int tap, int nb) {
    unsigned long long g = (unsigned long long)(wt + (size_t)tap * E + tid * 8);
    unsigned l = (unsigned)(size_t)(&smem[nb][tid * 8]);
#pragma unroll
    for (int e = 0; e < NCOPY; ++e) {
      asm volatile("global_load_async_to_lds_b128 %0, %1, off"
                   :: "v"(l), "v"(g) : "memory");
      g += 4096;   // 2048 bf16
      l += 4096;
    }
  };

  v8f acc[4];
#pragma unroll
  for (int j = 0; j < 4; ++j)
#pragma unroll
    for (int i = 0; i < 8; ++i) acc[j][i] = 0.0f;

  stage(0, 0);
  asm volatile("s_wait_asynccnt 0x0" ::: "memory");
  __syncthreads();

  for (int tap = 0; tap < 27; ++tap) {
    const int cur = tap & 1;
    if (tap + 1 < 27) stage(tap + 1, cur ^ 1);   // prefetch next tap while computing

    const int dz = tap / 9, dy = (tap / 3) % 3, dx = tap % 3;
    const int iz = oz * stride + dz - 1;
    const int iy = oy * stride + dy - 1;
    const int ix = ox * stride + dx - 1;
    const bool valid = ((unsigned)iz < (unsigned)Din) &&
                       ((unsigned)iy < (unsigned)Din) &&
                       ((unsigned)ix < (unsigned)Din);
    const __bf16* ap = inB + (size_t)((iz * Din + iy) * Din + ix) * CIN;
    const __bf16* wl = &smem[cur][0];

#pragma unroll
    for (int c0 = 0; c0 < CIN; c0 += 32) {
      // A fragment: lane<16 K={0..7,16..23}, lane>=16 K={8..15,24..31}
      V16 A;
      A.h[0] = zero8();
      A.h[1] = zero8();
      if (valid) {
        const __bf16* p = ap + c0 + hi * 8;
        A.h[0] = *(const v8bf*)(p);
        A.h[1] = *(const v8bf*)(p + 16);
      }
      // Preload all 4 B fragments from LDS, then issue 4 back-to-back WMMAs.
      // Column n = lane&15; lane<16 K={0..7,16..23}, lane>=16 K={8..15,24..31}.
      V16 Bf[4];
#pragma unroll
      for (int j = 0; j < 4; ++j) {
        const __bf16* q = wl + (size_t)(co0 + j * 16 + m) * CIN + c0 + hi * 8;
        Bf[j].h[0] = *(const v8bf*)(q);
        Bf[j].h[1] = *(const v8bf*)(q + 16);
      }
#pragma unroll
      for (int j = 0; j < 4; ++j) {
        acc[j] = __builtin_amdgcn_wmma_f32_16x16x32_bf16(false, A.v, false, Bf[j].v,
                                                         (short)0, acc[j], false, false);
      }
    }

    asm volatile("s_wait_asynccnt 0x0" ::: "memory");
    __syncthreads();
  }

  // Epilogue. C/D layout: lane<16 -> N=lane, M=r ; lane>=16 -> N=lane-16, M=8+r
  const int n  = lane & 15;
  const int mb = hi * 8;
  const float* maskB = mask + (size_t)b * NVo;
  float bs[4];
#pragma unroll
  for (int j = 0; j < 4; ++j) bs[j] = bias[co0 + j * 16 + n];

#pragma unroll
  for (int r = 0; r < 8; ++r) {
    const int vm = vt * 16 + mb + r;
    const float mval = maskB[vm];
#pragma unroll
    for (int j = 0; j < 4; ++j) {
      const int co = co0 + j * 16 + n;
      float f = (acc[j][r] + bs[j]) * mval;
      if (relu) f = fmaxf(f, 0.0f);
      if (res)  f += (float)res[((size_t)b * NVo + vm) * COUT + co];
      if (out)  out[((size_t)b * NVo + vm) * COUT + co] = (__bf16)f;
      if (outF) outF[((size_t)b * COUT + co) * NVo + vm] = f;
    }
  }
}

extern "C" void kernel_launch(void* const* d_in, const int* in_sizes, int n_in,
                              void* d_out, int out_size, void* d_ws, size_t ws_size,
                              hipStream_t stream) {
  const float* x = (const float*)d_in[0];
  const unsigned char* maskb = (const unsigned char*)d_in[1];
  const float* w0  = (const float*)d_in[2];
  const float* b0  = (const float*)d_in[3];
  const float* w00 = (const float*)d_in[4];
  const float* b00 = (const float*)d_in[5];
  const float* bw[12];
  const float* bb[12];
  for (int c = 0; c < 12; ++c) {
    int blk = c / 6, rem = c % 6, j = rem / 2, which = rem % 2;
    int base = 6 + blk * 12 + j * 4 + which * 2;
    bw[c] = (const float*)d_in[base];
    bb[c] = (const float*)d_in[base + 1];
  }
  const float* w1  = (const float*)d_in[30];
  const float* b1  = (const float*)d_in[31];
  const float* w10 = (const float*)d_in[32];
  const float* b10 = (const float*)d_in[33];
  const float* w2  = (const float*)d_in[34];
  const float* b2  = (const float*)d_in[35];
  const float* w20 = (const float*)d_in[36];
  const float* b20 = (const float*)d_in[37];

  // ---- workspace carve-out ----
  char* ws = (char*)d_ws;
  size_t off = 0;
  auto alloc = [&](size_t bytes) -> char* {
    char* p = ws + off;
    off += (bytes + 255) & ~(size_t)255;
    return p;
  };

  float* m0a = (float*)alloc((size_t)2 * 262144 * 4);
  float* m1a = (float*)alloc((size_t)2 * 32768 * 4);
  float* m2a = (float*)alloc((size_t)2 * 4096 * 4);
  float* m3a = (float*)alloc((size_t)2 * 512 * 4);

  __bf16* h0in  = (__bf16*)alloc((size_t)2 * 262144 * 32 * 2);
  __bf16* h0out = (__bf16*)alloc((size_t)2 * 262144 * 64 * 2);
  __bf16* A32 = (__bf16*)alloc((size_t)2 * 32768 * 128 * 2);
  __bf16* B32 = (__bf16*)alloc((size_t)2 * 32768 * 128 * 2);
  __bf16* C32 = (__bf16*)alloc((size_t)2 * 32768 * 128 * 2);
  __bf16* A16 = (__bf16*)alloc((size_t)2 * 4096 * 128 * 2);
  __bf16* B16 = (__bf16*)alloc((size_t)2 * 4096 * 128 * 2);
  __bf16* C16 = (__bf16*)alloc((size_t)2 * 4096 * 128 * 2);

  __bf16* wt0  = (__bf16*)alloc((size_t)27 * 64 * 32 * 2);
  __bf16* wt00 = (__bf16*)alloc((size_t)27 * 128 * 64 * 2);
  __bf16* wtB[12];
  for (int c = 0; c < 12; ++c) wtB[c] = (__bf16*)alloc((size_t)27 * 128 * 128 * 2);
  __bf16* wt1  = (__bf16*)alloc((size_t)27 * 128 * 128 * 2);
  __bf16* wt10 = (__bf16*)alloc((size_t)27 * 128 * 128 * 2);
  __bf16* wt2  = (__bf16*)alloc((size_t)27 * 128 * 128 * 2);
  __bf16* wt20 = (__bf16*)alloc((size_t)27 * 128 * 128 * 2);

  // ---- masks + input prep ----
  mask_from_bool<<<2048, 256, 0, stream>>>(maskb, m0a, 2 * 262144);
  pool_mask<<<256, 256, 0, stream>>>(m0a, m1a, 64);
  pool_mask<<<32, 256, 0, stream>>>(m1a, m2a, 32);
  pool_mask<<<4, 256, 0, stream>>>(m2a, m3a, 16);
  prep_input<<<2048, 256, 0, stream>>>(x, m0a, h0in);

  // ---- weight packing ----
  auto pack = [&](const float* w, __bf16* wt, int Cout, int Cin, int CinPad) {
    int total = 27 * Cout * CinPad;
    wt_pack<<<(total + 255) / 256, 256, 0, stream>>>(w, wt, Cout, Cin, CinPad, total);
  };
  pack(w0, wt0, 64, 3, 32);
  pack(w00, wt00, 128, 64, 64);
  for (int c = 0; c < 12; ++c) pack(bw[c], wtB[c], 128, 128, 128);
  pack(w1, wt1, 128, 128, 128);
  pack(w10, wt10, 128, 128, 128);
  pack(w2, wt2, 128, 128, 128);
  pack(w20, wt20, 128, 128, 128);

  // ---- conv chain ----
  auto convS = [&](const __bf16* in, const __bf16* wt, const float* bias, const float* msk,
                   __bf16* out, int Din, int Dout, int s, int relu) {
    int waves = 2 * ((Dout * Dout * Dout) / 16) * 1;    // COUT=64 -> 1 tile
    conv3d_wmma<32, 64><<<waves / 8, 256, 0, stream>>>(in, wt, bias, msk, nullptr, out,
                                                       nullptr, Din, Dout, s, relu);
  };
  auto convM = [&](const __bf16* in, const __bf16* wt, const float* bias, const float* msk,
                   __bf16* out, int Din, int Dout, int s, int relu) {
    int waves = 2 * ((Dout * Dout * Dout) / 16) * 2;    // COUT=128 -> 2 tiles
    conv3d_wmma<64, 128><<<waves / 8, 256, 0, stream>>>(in, wt, bias, msk, nullptr, out,
                                                        nullptr, Din, Dout, s, relu);
  };
  auto convL = [&](const __bf16* in, const __bf16* wt, const float* bias, const float* msk,
                   const __bf16* res, __bf16* out, float* outF,
                   int Din, int Dout, int s, int relu) {
    int waves = 2 * ((Dout * Dout * Dout) / 16) * 2;    // COUT=128 -> 2 tiles
    conv3d_wmma<128, 128><<<waves / 8, 256, 0, stream>>>(in, wt, bias, msk, res, out,
                                                         outF, Din, Dout, s, relu);
  };

  // stem
  convS(h0in, wt0, b0, m0a, h0out, 64, 64, 1, 0);
  convM(h0out, wt00, b00, m1a, A32, 64, 32, 2, 1);

  // block0 (32^3)
  __bf16 *hc = A32, *tp = B32, *hn = C32;
  for (int j = 0; j < 3; ++j) {
    int c = j * 2;
    convL(hc, wtB[c],     bb[c],     m1a, nullptr, tp, nullptr, 32, 32, 1, 1);
    convL(tp, wtB[c + 1], bb[c + 1], m1a, hc,      hn, nullptr, 32, 32, 1, 0);
    __bf16* sw = hc; hc = hn; hn = sw;
  }
  convL(hc, wt1, b1, m1a, nullptr, tp, nullptr, 32, 32, 1, 0);
  convL(tp, wt10, b10, m2a, nullptr, A16, nullptr, 32, 16, 2, 1);

  // block1 (16^3)
  __bf16 *h2 = A16, *t2 = B16, *n2 = C16;
  for (int j = 0; j < 3; ++j) {
    int c = 6 + j * 2;
    convL(h2, wtB[c],     bb[c],     m2a, nullptr, t2, nullptr, 16, 16, 1, 1);
    convL(t2, wtB[c + 1], bb[c + 1], m2a, h2,      n2, nullptr, 16, 16, 1, 0);
    __bf16* sw = h2; h2 = n2; n2 = sw;
  }
  convL(h2, wt2, b2, m2a, nullptr, t2, nullptr, 16, 16, 1, 0);
  // final: write f32 NCDHW into d_out
  convL(t2, wt20, b20, m3a, nullptr, nullptr, (float*)d_out, 16, 8, 2, 0);

  (void)in_sizes; (void)n_in; (void)out_size; (void)ws_size;
}
